// DeepSeekV3MoE_58299886076020
// MI455X (gfx1250) — matrix operand order repaired and checked by reading, not compiled, not run on hardware
//
#include <hip/hip_runtime.h>
#include <math.h>

// ---------------------------------------------------------------------------
// DeepSeek-V3 MoE for gfx1250 (MI455X), bf16 WMMA path.
// Sparse top-2 routed experts + shared expert, fused up/down FFN per tile.
// All GEMM staging via global_load_async_to_lds_b128 (ASYNCcnt), double-
// buffered with ONE barrier per K-step. Weights pre-converted to bf16 in ws.
// LDS double-buffer selected via integer offsets (keeps addrspace(3) -> ds_*).
// ---------------------------------------------------------------------------

typedef __bf16 bf16_t;
typedef __bf16 v16bf __attribute__((ext_vector_type(16)));
typedef __bf16 v4bf  __attribute__((ext_vector_type(4)));
typedef float  v8f   __attribute__((ext_vector_type(8)));

#define T_TOK 4096   // B*S tokens
#define HDIM  2048   // hidden size
#define FDIM  1408   // intermediate size
#define NE    8      // routed experts
#define MT    32     // token tile (rows per block)
#define NT    128    // N tile (columns per block)
#define KT    64     // K elements staged per step (2 WMMA k-slices)
#define XPAD  8      // x/w LDS tile row pad (stride 72 bf16 = 144B, 16B-mult)
#define APAD  8      // act LDS row pad (stride 1416 bf16 = 2832B, 16B-mult)

static constexpr int ACT_STRIDE = FDIM + APAD;  // 1416
static constexpr int XT_STRIDE  = KT + XPAD;    // 72
// LDS element-offset layout (bf16 elements):
static constexpr int BUF_ELEMS = (MT + 2 * NT) * XT_STRIDE;   // 20,736 per buffer
static constexpr int XT_OFF    = MT * ACT_STRIDE;             // buffer0 xT
static constexpr int WA_OFF    = XT_OFF + MT * XT_STRIDE;     // buffer0 wA
static constexpr int WB_OFF    = XT_OFF + (MT + NT) * XT_STRIDE; // buffer0 wB
static constexpr int META_OFF  = XT_OFF + 2 * BUF_ELEMS;      // ints after buffers
static constexpr size_t SMEM_TOTAL =
    (size_t)META_OFF * sizeof(bf16_t) + MT * (2 * sizeof(int) + sizeof(float));

__device__ __forceinline__ float fast_sigmoid(float x) {
    return __builtin_amdgcn_rcpf(1.0f + __expf(-x));   // v_exp_f32 + v_rcp_f32
}

__device__ __forceinline__ void async_b128(unsigned ldsoff, const void* g) {
    asm volatile("global_load_async_to_lds_b128 %0, %1, off"
                 :: "v"(ldsoff), "v"(g) : "memory");
}
__device__ __forceinline__ void wait_async0() {
    asm volatile("s_wait_asynccnt 0x0" ::: "memory");
}

// A fragment (16x32 bf16): lane<16 -> K 0-7 & 16-23; lane>=16 -> K 8-15 & 24-31
__device__ __forceinline__ v16bf load_a_frag(const bf16_t* rowp, int lhalf) {
    v16bf a;
    int kb = lhalf * 8;
#pragma unroll
    for (int j = 0; j < 8; ++j) {
        a[j]     = rowp[kb + j];
        a[8 + j] = rowp[16 + kb + j];
    }
    return a;
}

// B fragment (32x16 bf16): lane<16 -> K 0-7 & 16-23; lane>=16 -> K 8-15 & 24-31; N = lane&15
__device__ __forceinline__ v16bf load_b_frag(const bf16_t* colp, int lhalf) {
    v16bf b;
    int kb = lhalf * 8;
#pragma unroll
    for (int j = 0; j < 8; ++j) {
        b[j]     = colp[kb + j];
        b[8 + j] = colp[16 + kb + j];
    }
    return b;
}

// ---------------------------------------------------------------------------
// Kernel 1a: x (fp32) -> xb (bf16); also zero expert counters.
// ---------------------------------------------------------------------------
__global__ __launch_bounds__(256) void k_convert_x(const float* __restrict__ x,
                                                   bf16_t* __restrict__ xb,
                                                   int* __restrict__ cnt) {
    if (blockIdx.x == 0 && threadIdx.x < NE) cnt[threadIdx.x] = 0;
    size_t i = ((size_t)blockIdx.x * 256 + threadIdx.x) * 4;
    float4 v = *(const float4*)(x + i);
    v4bf o;
    o[0] = (bf16_t)v.x; o[1] = (bf16_t)v.y; o[2] = (bf16_t)v.z; o[3] = (bf16_t)v.w;
    *(v4bf*)(xb + i) = o;
}

// Kernel 1b: generic fp32 -> bf16 converter (exact grids, no bounds check)
__global__ __launch_bounds__(256) void k_cvt(const float* __restrict__ s,
                                             bf16_t* __restrict__ d) {
    size_t i = ((size_t)blockIdx.x * 256 + threadIdx.x) * 4;
    float4 v = *(const float4*)(s + i);
    v4bf o;
    o[0] = (bf16_t)v.x; o[1] = (bf16_t)v.y; o[2] = (bf16_t)v.z; o[3] = (bf16_t)v.w;
    *(v4bf*)(d + i) = o;
}

// ---------------------------------------------------------------------------
// Kernel 2: gating. One wave per token: sigmoid(x @ gate_w^T), bias-adjusted
// top-2, normalized gates; atomic-append (tok, gate, slot) to expert lists.
// ---------------------------------------------------------------------------
__global__ __launch_bounds__(256) void k_gate(const float* __restrict__ x,
                                              const float* __restrict__ gw,
                                              const float* __restrict__ bias,
                                              int* __restrict__ cnt,
                                              int* __restrict__ tokL,
                                              float* __restrict__ gateL,
                                              int* __restrict__ slotL) {
    const int wave = threadIdx.x >> 5;
    const int lane = threadIdx.x & 31;
    const int t = blockIdx.x * 8 + wave;

    float acc[NE];
#pragma unroll
    for (int e = 0; e < NE; ++e) acc[e] = 0.0f;

    const float* xr = x + (size_t)t * HDIM;
    for (int h = lane; h < HDIM; h += 32) {
        float xv = xr[h];
#pragma unroll
        for (int e = 0; e < NE; ++e) acc[e] += xv * gw[e * HDIM + h];
    }
#pragma unroll
    for (int e = 0; e < NE; ++e) {
#pragma unroll
        for (int off = 16; off > 0; off >>= 1)
            acc[e] += __shfl_xor(acc[e], off, 32);
    }

    if (lane == 0) {
        float s[NE];
#pragma unroll
        for (int e = 0; e < NE; ++e) s[e] = fast_sigmoid(acc[e]);
        int i1 = 0; float b1 = -1e30f;
#pragma unroll
        for (int e = 0; e < NE; ++e) {
            float r = s[e] + bias[e];
            if (r > b1) { b1 = r; i1 = e; }
        }
        int i2 = (i1 == 0) ? 1 : 0; float b2 = -1e30f;
#pragma unroll
        for (int e = 0; e < NE; ++e) {
            if (e == i1) continue;
            float r = s[e] + bias[e];
            if (r > b2) { b2 = r; i2 = e; }
        }
        float inv = __builtin_amdgcn_rcpf(s[i1] + s[i2]);
        float g1 = s[i1] * inv, g2 = s[i2] * inv;
        int p = atomicAdd(&cnt[i1], 1);
        tokL[i1 * T_TOK + p] = t; gateL[i1 * T_TOK + p] = g1; slotL[i1 * T_TOK + p] = 0;
        p = atomicAdd(&cnt[i2], 1);
        tokL[i2 * T_TOK + p] = t; gateL[i2 * T_TOK + p] = g2; slotL[i2 * T_TOK + p] = 1;
    }
}

// ---------------------------------------------------------------------------
// Kernels 3/4: fused SwiGLU FFN per (expert, 32-token tile), bf16 WMMA.
// Weights are pre-converted bf16; "expert NE" holds the shared expert.
// Double-buffered async staging, one barrier per K-step.
// ---------------------------------------------------------------------------
template <bool SHARED>
__global__ __launch_bounds__(256) void k_moe_ffn(const bf16_t* __restrict__ xb,
                                                 const bf16_t* __restrict__ w1,
                                                 const bf16_t* __restrict__ w3,
                                                 const bf16_t* __restrict__ w2,
                                                 const int* __restrict__ cnt,
                                                 const int* __restrict__ tokL,
                                                 const float* __restrict__ gateL,
                                                 const int* __restrict__ slotL,
                                                 float* __restrict__ outp) {
    extern __shared__ bf16_t sm[];
    int*   tokS  = (int*)(sm + META_OFF);
    float* gateS = (float*)(tokS + MT);
    int*   slotS = (int*)(gateS + MT);

    const int e = SHARED ? NE : blockIdx.y;
    const int tileBase = blockIdx.x * MT;
    const int cntE = SHARED ? T_TOK : cnt[e];
    if (tileBase >= cntE) return;   // uniform early-exit (before any barrier)

    const int tid   = threadIdx.x;
    const int lane  = tid & 31;
    const int wave  = tid >> 5;
    const int lhalf = lane >> 4;
    const int l15   = lane & 15;
    const unsigned smA = (unsigned)(uintptr_t)sm;   // LDS byte base for async asm

    if (tid < MT) {
        int row = tileBase + tid;
        if (SHARED) {
            tokS[tid] = row; gateS[tid] = 1.0f; slotS[tid] = 0;
        } else if (row < cntE) {
            tokS[tid] = tokL[e * T_TOK + row];
            gateS[tid] = gateL[e * T_TOK + row];
            slotS[tid] = slotL[e * T_TOK + row];
        } else {
            tokS[tid] = 0; gateS[tid] = 0.0f; slotS[tid] = 0;
        }
    }
    __syncthreads();

    const bf16_t* w1e = w1 + (size_t)e * FDIM * HDIM;
    const bf16_t* w3e = w3 + (size_t)e * FDIM * HDIM;
    const bf16_t* w2e = w2 + (size_t)e * HDIM * FDIM;

    // per-thread staging coordinates
    const int xr = tid >> 3, xk = (tid & 7) * 8;          // x tile: 1 x b128
    const int wr = tid >> 1, wk = (tid & 1) * 32;         // w tile: 4 x b128
    const unsigned ldsX = smA + (unsigned)(XT_OFF + xr * XT_STRIDE + xk) * 2;
    const unsigned ldsA = smA + (unsigned)(WA_OFF + wr * XT_STRIDE + wk) * 2;
    const unsigned ldsB = smA + (unsigned)(WB_OFF + wr * XT_STRIDE + wk) * 2;
    const unsigned bufB = (unsigned)BUF_ELEMS * 2;        // byte stride buf0->buf1
    const size_t xrow = (size_t)tokS[xr] * HDIM + xk;

    // ---------------- UP projection ----------------
    for (int nt = 0; nt < FDIM; nt += NT) {
        v8f c1[2] = {{}, {}};
        v8f c3[2] = {{}, {}};
        const bf16_t* w1n = w1e + (size_t)nt * HDIM + (size_t)wr * HDIM + wk;
        const bf16_t* w3n = w3e + (size_t)nt * HDIM + (size_t)wr * HDIM + wk;
        int p = 0;
        {   // prologue: stage kt=0 into buf0
            async_b128(ldsX, xb + xrow);
#pragma unroll
            for (int j = 0; j < 32; j += 8) {
                async_b128(ldsA + j * 2, w1n + j);
                async_b128(ldsB + j * 2, w3n + j);
            }
        }
        wait_async0();
        __syncthreads();
        for (int kt = 0; kt < HDIM; kt += KT) {
            int kn = kt + KT;
            if (kn < HDIM) {   // stage next K-step into the other buffer
                unsigned bo = (p ^ 1) ? bufB : 0u;
                async_b128(ldsX + bo, xb + xrow + kn);
#pragma unroll
                for (int j = 0; j < 32; j += 8) {
                    async_b128(ldsA + bo + j * 2, w1n + kn + j);
                    async_b128(ldsB + bo + j * 2, w3n + kn + j);
                }
            }
            const int bo = p * BUF_ELEMS;
#pragma unroll
            for (int ks = 0; ks < KT; ks += 32) {
                v16bf a0 = load_a_frag(sm + XT_OFF + bo + l15 * XT_STRIDE + ks, lhalf);
                v16bf a1 = load_a_frag(sm + XT_OFF + bo + (16 + l15) * XT_STRIDE + ks, lhalf);
                v16bf b1f = load_b_frag(sm + WA_OFF + bo + (wave * 16 + l15) * XT_STRIDE + ks, lhalf);
                v16bf b3f = load_b_frag(sm + WB_OFF + bo + (wave * 16 + l15) * XT_STRIDE + ks, lhalf);
                c1[0] = __builtin_amdgcn_wmma_f32_16x16x32_bf16(false, a0, false, b1f,
                                                                (short)0, c1[0], false, false);
                c1[1] = __builtin_amdgcn_wmma_f32_16x16x32_bf16(false, a1, false, b1f,
                                                                (short)0, c1[1], false, false);
                c3[0] = __builtin_amdgcn_wmma_f32_16x16x32_bf16(false, a0, false, b3f,
                                                                (short)0, c3[0], false, false);
                c3[1] = __builtin_amdgcn_wmma_f32_16x16x32_bf16(false, a1, false, b3f,
                                                                (short)0, c3[1], false, false);
            }
            wait_async0();     // own next-step loads landed (hidden behind WMMAs)
            __syncthreads();   // everyone done reading buf[p]; everyone's loads landed
            p ^= 1;
        }
        // epilogue: act = gate * silu(h1) * h3 -> LDS (bf16)
#pragma unroll
        for (int ms = 0; ms < 2; ++ms) {
#pragma unroll
            for (int r = 0; r < 8; ++r) {
                int m = ms * 16 + lhalf * 8 + r;   // C layout: lanes<16 M=r, else M=8+r
                float a = c1[ms][r];
                float b = c3[ms][r];
                float v = gateS[m] * (a * fast_sigmoid(a)) * b;
                sm[m * ACT_STRIDE + nt + wave * 16 + l15] = (bf16_t)v;
            }
        }
    }
    __syncthreads();

    // ---------------- DOWN projection ----------------
    for (int ht = 0; ht < HDIM; ht += NT) {
        v8f oc[2] = {{}, {}};
        const bf16_t* w2n = w2e + (size_t)ht * FDIM + (size_t)wr * FDIM + wk;
        int p = 0;
        {   // prologue
#pragma unroll
            for (int j = 0; j < 32; j += 8) async_b128(ldsA + j * 2, w2n + j);
        }
        wait_async0();
        __syncthreads();
        for (int kt = 0; kt < FDIM; kt += KT) {
            int kn = kt + KT;
            if (kn < FDIM) {
                unsigned bo = (p ^ 1) ? bufB : 0u;
#pragma unroll
                for (int j = 0; j < 32; j += 8) async_b128(ldsA + bo + j * 2, w2n + kn + j);
            }
            const int bo = p * BUF_ELEMS;
#pragma unroll
            for (int ks = 0; ks < KT; ks += 32) {
                v16bf a0 = load_a_frag(sm + l15 * ACT_STRIDE + kt + ks, lhalf);
                v16bf a1 = load_a_frag(sm + (16 + l15) * ACT_STRIDE + kt + ks, lhalf);
                v16bf bf = load_b_frag(sm + WA_OFF + bo + (wave * 16 + l15) * XT_STRIDE + ks, lhalf);
                oc[0] = __builtin_amdgcn_wmma_f32_16x16x32_bf16(false, a0, false, bf,
                                                                (short)0, oc[0], false, false);
                oc[1] = __builtin_amdgcn_wmma_f32_16x16x32_bf16(false, a1, false, bf,
                                                                (short)0, oc[1], false, false);
            }
            wait_async0();
            __syncthreads();
            p ^= 1;
        }
        // epilogue: write this wave's 16 columns of the output tile
        const int col = ht + wave * 16 + l15;
#pragma unroll
        for (int ms = 0; ms < 2; ++ms) {
#pragma unroll
            for (int r = 0; r < 8; ++r) {
                int m = ms * 16 + lhalf * 8 + r;
                int row = tileBase + m;
                if (SHARED) {
                    outp[(size_t)tokS[m] * HDIM + col] = oc[ms][r];
                } else if (row < cntE) {
                    outp[((size_t)slotS[m] * T_TOK + tokS[m]) * HDIM + col] = oc[ms][r];
                }
            }
        }
    }
}

// ---------------------------------------------------------------------------
// Kernel 5: out = shared(out) + partial[0] + partial[1]
// ---------------------------------------------------------------------------
__global__ __launch_bounds__(256) void k_combine(float* __restrict__ out,
                                                 const float* __restrict__ part) {
    size_t i = ((size_t)blockIdx.x * 256 + threadIdx.x) * 4;
    float4 o  = *(float4*)(out + i);
    float4 p0 = *(const float4*)(part + i);
    float4 p1 = *(const float4*)(part + (size_t)T_TOK * HDIM + i);
    o.x += p0.x + p1.x;
    o.y += p0.y + p1.y;
    o.z += p0.z + p1.z;
    o.w += p0.w + p1.w;
    *(float4*)(out + i) = o;
}

// ---------------------------------------------------------------------------
// Host launcher
// ---------------------------------------------------------------------------
extern "C" void kernel_launch(void* const* d_in, const int* in_sizes, int n_in,
                              void* d_out, int out_size, void* d_ws, size_t ws_size,
                              hipStream_t stream) {
    const float* x      = (const float*)d_in[0];   // [T, H]
    const float* gate_w = (const float*)d_in[1];   // [E, H]
    const float* bias   = (const float*)d_in[2];   // [E]
    const float* ws1    = (const float*)d_in[3];   // [F, H]
    const float* ws2    = (const float*)d_in[4];   // [H, F]
    const float* ws3    = (const float*)d_in[5];   // [F, H]
    const float* we1    = (const float*)d_in[6];   // [E, F, H]
    const float* we2    = (const float*)d_in[7];   // [E, H, F]
    const float* we3    = (const float*)d_in[8];   // [E, F, H]
    float* out = (float*)d_out;

    constexpr size_t WEXP = (size_t)FDIM * HDIM;        // one expert's matrix
    // workspace layout
    char* w = (char*)d_ws;
    size_t off = 0;
    bf16_t* xb  = (bf16_t*)(w + off); off += (size_t)T_TOK * HDIM * sizeof(bf16_t);
    bf16_t* w1b = (bf16_t*)(w + off); off += (NE + 1) * WEXP * sizeof(bf16_t);
    bf16_t* w3b = (bf16_t*)(w + off); off += (NE + 1) * WEXP * sizeof(bf16_t);
    bf16_t* w2b = (bf16_t*)(w + off); off += (NE + 1) * WEXP * sizeof(bf16_t);
    int*    cnt  = (int*)(w + off);   off += 256;
    int*    tokL = (int*)(w + off);   off += (size_t)NE * T_TOK * sizeof(int);
    float*  gatL = (float*)(w + off); off += (size_t)NE * T_TOK * sizeof(float);
    int*    sloL = (int*)(w + off);   off += (size_t)NE * T_TOK * sizeof(int);
    float*  part = (float*)(w + off); off += (size_t)2 * T_TOK * HDIM * sizeof(float);
    (void)ws_size; (void)in_sizes; (void)n_in; (void)out_size;

    // 1) convert x + all weights to bf16 (shared expert packed as expert NE)
    k_convert_x<<<(T_TOK * HDIM) / 1024, 256, 0, stream>>>(x, xb, cnt);
    k_cvt<<<(NE * WEXP) / 1024, 256, 0, stream>>>(we1, w1b);
    k_cvt<<<WEXP / 1024, 256, 0, stream>>>(ws1, w1b + NE * WEXP);
    k_cvt<<<(NE * WEXP) / 1024, 256, 0, stream>>>(we3, w3b);
    k_cvt<<<WEXP / 1024, 256, 0, stream>>>(ws3, w3b + NE * WEXP);
    k_cvt<<<(NE * WEXP) / 1024, 256, 0, stream>>>(we2, w2b);
    k_cvt<<<WEXP / 1024, 256, 0, stream>>>(ws2, w2b + NE * WEXP);

    // 2) gating / top-2 routing
    k_gate<<<T_TOK / 8, 256, 0, stream>>>(x, gate_w, bias, cnt, tokL, gatL, sloL);

    // 3) shared expert: writes d_out directly
    k_moe_ffn<true><<<dim3(T_TOK / MT, 1), 256, SMEM_TOTAL, stream>>>(
        xb, w1b, w3b, w2b, cnt, tokL, gatL, sloL, out);

    // 4) routed experts (gathered, gate-scaled): writes per-slot partials
    k_moe_ffn<false><<<dim3(T_TOK / MT, NE), 256, SMEM_TOTAL, stream>>>(
        xb, w1b, w3b, w2b, cnt, tokL, gatL, sloL, part);

    // 5) combine: out = shared + partial0 + partial1
    k_combine<<<(T_TOK * HDIM) / 1024, 256, 0, stream>>>(out, part);
}
